// Predictor_16252156248433
// MI455X (gfx1250) — compile-verified, matrix operand order checked by reading, not run on hardware
//
#include <hip/hip_runtime.h>
#include <hip/hip_bf16.h>

#define DIMD   768
#define HIDD   3072
#define NHEADD 12
#define HDIMD  64
#define BQ     2
#define MQ     4
#define NTT    64
#define LKK    512
#define TFULL  768
#define NDROP  256
#define SEQL   576            // LKK + NTT
#define NSEQ   8              // BQ * MQ
#define ROWS   (NSEQ * SEQL)  // 4608

typedef __attribute__((ext_vector_type(16))) __bf16 v16bf;
typedef __attribute__((ext_vector_type(8)))  float  v8f;

union FragAB { v16bf v; unsigned int u[8]; };

__device__ __forceinline__ unsigned short f32_to_bf16(float f) {
  unsigned int u = __float_as_uint(f);
  u += 0x7FFFu + ((u >> 16) & 1u);   // round-to-nearest-even
  return (unsigned short)(u >> 16);
}

// ---------------------------------------------------------------------------
// Generic bf16 WMMA GEMM: C[M,N] = A[M,K] @ B[K,N] (+bias)(+resid)(+GELU)
// A: bf16 row-major (M x K). BT: bf16 "B transposed" row-major (N x K).
// One wave (32 lanes) computes a 16(M) x 64(N) strip; K-loop in steps of 32.
// A-fragment layout (16-bit A 16x32): lane = half*16 + m; vgpr j holds
// K = (j<4 ? 2j : 16+2(j-4)) + lo + 8*half.  B uses symmetric per-column layout.
// C/D layout: vgpr j = row (j + 8*half), lane&15 = col.
// ---------------------------------------------------------------------------
__global__ __launch_bounds__(32)
void gemm_bf16_wmma(const unsigned short* __restrict__ A,
                    const unsigned short* __restrict__ BT,
                    const float* __restrict__ bias,
                    const float* resid,            // may alias outF (in-place residual)
                    float* outF,                   // fp32 out (nullable)
                    unsigned short* __restrict__ outB, // bf16 out (nullable)
                    int M, int N, int K, int fuse_gelu)
{
  const int lane = threadIdx.x & 31;
  const int half = lane >> 4;
  const int lm   = lane & 15;
  const int tN   = blockIdx.x;   // 64-wide tiles
  const int tM   = blockIdx.y;   // 16-high tiles
  (void)M;

  const v8f zero = {0.f,0.f,0.f,0.f,0.f,0.f,0.f,0.f};
  v8f acc[4];
  #pragma unroll
  for (int s = 0; s < 4; ++s) acc[s] = zero;

  const unsigned short* arow = A  + (size_t)(tM*16 + lm) * K;
  const unsigned short* brow[4];
  #pragma unroll
  for (int s = 0; s < 4; ++s) brow[s] = BT + (size_t)(tN*64 + s*16 + lm) * K;

  for (int k0 = 0; k0 < K; k0 += 32) {
    FragAB fa, fb0, fb1, fb2, fb3;
    #pragma unroll
    for (int j = 0; j < 8; ++j) {
      const int kj = k0 + ((j < 4) ? (2*j) : (16 + 2*(j-4))) + 8*half;
      fa.u[j]  = *(const unsigned int*)(arow    + kj);
      fb0.u[j] = *(const unsigned int*)(brow[0] + kj);
      fb1.u[j] = *(const unsigned int*)(brow[1] + kj);
      fb2.u[j] = *(const unsigned int*)(brow[2] + kj);
      fb3.u[j] = *(const unsigned int*)(brow[3] + kj);
    }
    acc[0] = __builtin_amdgcn_wmma_f32_16x16x32_bf16(false, fa.v, false, fb0.v, (short)0, acc[0], false, false);
    acc[1] = __builtin_amdgcn_wmma_f32_16x16x32_bf16(false, fa.v, false, fb1.v, (short)0, acc[1], false, false);
    acc[2] = __builtin_amdgcn_wmma_f32_16x16x32_bf16(false, fa.v, false, fb2.v, (short)0, acc[2], false, false);
    acc[3] = __builtin_amdgcn_wmma_f32_16x16x32_bf16(false, fa.v, false, fb3.v, (short)0, acc[3], false, false);
  }

  #pragma unroll
  for (int j = 0; j < 8; ++j) {
    const int row = tM*16 + 8*half + j;
    #pragma unroll
    for (int s = 0; s < 4; ++s) {
      const int col = tN*64 + s*16 + lm;
      float v = acc[s][j];
      if (bias)  v += bias[col];
      if (resid) v += resid[(size_t)row * N + col];
      if (fuse_gelu) v = 0.5f * v * (1.0f + erff(v * 0.70710678118654752f));
      if (outF) outF[(size_t)row * N + col] = v;
      if (outB) outB[(size_t)row * N + col] = f32_to_bf16(v);
    }
  }
}

// fp32 (K x N) row-major -> bf16 transposed (N x K) row-major
__global__ void cvt_transpose(const float* __restrict__ W,
                              unsigned short* __restrict__ WT, int K, int N)
{
  int idx = blockIdx.x * blockDim.x + threadIdx.x;
  if (idx >= K * N) return;
  int n = idx / K, k = idx % K;
  WT[idx] = f32_to_bf16(W[(size_t)k * N + n]);
}

// keep_idx = indices of [0,768) not present in sorted ctx_mask (256 entries)
__global__ void build_keep(const int* __restrict__ ctx_mask, int* __restrict__ keep_idx)
{
  if (blockIdx.x == 0 && threadIdx.x == 0) {
    int p = 0, j = 0;
    for (int t = 0; t < TFULL; ++t) {
      if (p < NDROP && ctx_mask[p] == t) { ++p; continue; }
      if (j < LKK) keep_idx[j++] = t;
    }
  }
}

// xctx[b,l,d] = bf16( ctx_feature[b,d,l] + pos_embed[keep_idx[l], d] )
__global__ void build_xctx(const float* __restrict__ cf, const float* __restrict__ pe,
                           const int* __restrict__ keep_idx, unsigned short* __restrict__ xctx)
{
  int idx = blockIdx.x * blockDim.x + threadIdx.x;
  if (idx >= BQ * LKK * DIMD) return;
  int d = idx % DIMD;
  int l = (idx / DIMD) % LKK;
  int b = idx / (DIMD * LKK);
  float v = cf[((size_t)b * DIMD + d) * LKK + l] + pe[(size_t)keep_idx[l] * DIMD + d];
  xctx[idx] = f32_to_bf16(v);
}

// broadcast embedded ctx (B,Lk,D) -> h rows for all M query groups
__global__ void replicate_ctx(const float* __restrict__ y, float* __restrict__ h)
{
  int idx = blockIdx.x * blockDim.x + threadIdx.x;
  if (idx >= NSEQ * LKK * DIMD) return;
  int d = idx % DIMD;
  int l = (idx / DIMD) % LKK;
  int bm = idx / (DIMD * LKK);
  int b = bm / MQ;
  h[((size_t)bm * SEQL + l) * DIMD + d] = y[((size_t)b * LKK + l) * DIMD + d];
}

// h[bm, 512+n, d] = mask_tok[d] + (valid ? pe[clip(idx)] : 0)
__global__ void build_tgt(const float* __restrict__ mask_tok, const float* __restrict__ pe,
                          const int* __restrict__ tgt_mask, float* __restrict__ h)
{
  int idx = blockIdx.x * blockDim.x + threadIdx.x;
  if (idx >= NSEQ * NTT * DIMD) return;
  int d = idx % DIMD;
  int n = (idx / DIMD) % NTT;
  int bm = idx / (DIMD * NTT);
  int t = tgt_mask[bm * NTT + n];
  bool valid = (t != -1) && (t < TFULL);
  int tc = min(max(t, 0), TFULL - 1);
  float v = mask_tok[d] + (valid ? pe[(size_t)tc * DIMD + d] : 0.0f);
  h[((size_t)bm * SEQL + LKK + n) * DIMD + d] = v;
}

// LayerNorm over 768, write bf16 (GEMM operand A)
__global__ __launch_bounds__(256)
void ln_rows_bf16(const float* __restrict__ x, const float* __restrict__ g,
                  const float* __restrict__ b, unsigned short* __restrict__ y)
{
  __shared__ float sred[256];
  const int row = blockIdx.x, tid = threadIdx.x;
  const float* xr = x + (size_t)row * DIMD;
  float v0 = xr[tid], v1 = xr[tid + 256], v2 = xr[tid + 512];
  sred[tid] = v0 + v1 + v2; __syncthreads();
  for (int s = 128; s > 0; s >>= 1) { if (tid < s) sred[tid] += sred[tid + s]; __syncthreads(); }
  const float mean = sred[0] * (1.0f / DIMD); __syncthreads();
  const float d0 = v0 - mean, d1 = v1 - mean, d2 = v2 - mean;
  sred[tid] = d0*d0 + d1*d1 + d2*d2; __syncthreads();
  for (int s = 128; s > 0; s >>= 1) { if (tid < s) sred[tid] += sred[tid + s]; __syncthreads(); }
  const float rstd = rsqrtf(sred[0] * (1.0f / DIMD) + 1e-5f);
  unsigned short* yr = y + (size_t)row * DIMD;
  yr[tid]       = f32_to_bf16(d0 * rstd * g[tid]       + b[tid]);
  yr[tid + 256] = f32_to_bf16(d1 * rstd * g[tid + 256] + b[tid + 256]);
  yr[tid + 512] = f32_to_bf16(d2 * rstd * g[tid + 512] + b[tid + 512]);
}

// Final LN on the last 64 tokens of each sequence, fp32 out (8,64,768)
__global__ __launch_bounds__(256)
void final_ln(const float* __restrict__ x, const float* __restrict__ g,
              const float* __restrict__ b, float* __restrict__ out)
{
  __shared__ float sred[256];
  const int bm = blockIdx.x / NTT, n = blockIdx.x % NTT, tid = threadIdx.x;
  const float* xr = x + ((size_t)bm * SEQL + LKK + n) * DIMD;
  float v0 = xr[tid], v1 = xr[tid + 256], v2 = xr[tid + 512];
  sred[tid] = v0 + v1 + v2; __syncthreads();
  for (int s = 128; s > 0; s >>= 1) { if (tid < s) sred[tid] += sred[tid + s]; __syncthreads(); }
  const float mean = sred[0] * (1.0f / DIMD); __syncthreads();
  const float d0 = v0 - mean, d1 = v1 - mean, d2 = v2 - mean;
  sred[tid] = d0*d0 + d1*d1 + d2*d2; __syncthreads();
  for (int s = 128; s > 0; s >>= 1) { if (tid < s) sred[tid] += sred[tid + s]; __syncthreads(); }
  const float rstd = rsqrtf(sred[0] * (1.0f / DIMD) + 1e-5f);
  float* orow = out + ((size_t)bm * NTT + n) * DIMD;
  orow[tid]       = d0 * rstd * g[tid]       + b[tid];
  orow[tid + 256] = d1 * rstd * g[tid + 256] + b[tid + 256];
  orow[tid + 512] = d2 * rstd * g[tid + 512] + b[tid + 512];
}

// Softmax attention, one block (64 threads) per (bm, head, q). key_mask is
// all-True in this config, so it's a plain full softmax over 576 keys.
// qkv layout: [row = bm*576+t][ j*768 + head*64 + d ], j = 0(q),1(k),2(v).
__global__ __launch_bounds__(64)
void attention(const float* __restrict__ qkv, unsigned short* __restrict__ aout)
{
  __shared__ float qs[HDIMD];
  __shared__ float sc[SEQL];
  __shared__ float red[64];
  const int q = blockIdx.x, hd = blockIdx.y, bm = blockIdx.z, tid = threadIdx.x;
  const size_t rs = 3 * DIMD;
  const float* base = qkv + (size_t)bm * SEQL * rs;

  qs[tid] = base[(size_t)q * rs + hd * HDIMD + tid] * 0.125f; // 1/sqrt(64)
  __syncthreads();

  float loc[SEQL / 64];
  float lmax = -3.0e38f;
  #pragma unroll
  for (int i = 0; i < SEQL / 64; ++i) {
    const int t = tid + 64 * i;
    const float* krow = base + (size_t)t * rs + DIMD + hd * HDIMD;
    float dot = 0.f;
    #pragma unroll
    for (int d = 0; d < HDIMD; ++d) dot += qs[d] * krow[d];
    loc[i] = dot;
    lmax = fmaxf(lmax, dot);
  }
  red[tid] = lmax; __syncthreads();
  for (int s = 32; s > 0; s >>= 1) { if (tid < s) red[tid] = fmaxf(red[tid], red[tid + s]); __syncthreads(); }
  const float m = red[0]; __syncthreads();

  float lsum = 0.f;
  #pragma unroll
  for (int i = 0; i < SEQL / 64; ++i) {
    const float e = expf(loc[i] - m);
    sc[tid + 64 * i] = e;
    lsum += e;
  }
  red[tid] = lsum; __syncthreads();
  for (int s = 32; s > 0; s >>= 1) { if (tid < s) red[tid] += red[tid + s]; __syncthreads(); }
  const float inv = 1.0f / red[0]; __syncthreads();

  float acc = 0.f;
  for (int t = 0; t < SEQL; ++t)
    acc += sc[t] * base[(size_t)t * rs + 2 * DIMD + hd * HDIMD + tid];
  aout[((size_t)bm * SEQL + q) * DIMD + hd * HDIMD + tid] = f32_to_bf16(acc * inv);
}

extern "C" void kernel_launch(void* const* d_in, const int* in_sizes, int n_in,
                              void* d_out, int out_size, void* d_ws, size_t ws_size,
                              hipStream_t stream)
{
  const float* ctx_feature = (const float*)d_in[0];
  const float* embed_w     = (const float*)d_in[1];
  const float* embed_b     = (const float*)d_in[2];
  const float* mask_tok    = (const float*)d_in[3];
  const float* pos_embed   = (const float*)d_in[4];
  const float* ln_w        = (const float*)d_in[5];
  const float* ln_b        = (const float*)d_in[6];
  const float* Wqkv        = (const float*)d_in[7];
  const float* bqkv        = (const float*)d_in[8];
  const float* Wo          = (const float*)d_in[9];
  const float* bo          = (const float*)d_in[10];
  const float* ln1w        = (const float*)d_in[11];
  const float* ln1b        = (const float*)d_in[12];
  const float* ln2w        = (const float*)d_in[13];
  const float* ln2b        = (const float*)d_in[14];
  const float* W1          = (const float*)d_in[15];
  const float* b1          = (const float*)d_in[16];
  const float* W2          = (const float*)d_in[17];
  const float* b2          = (const float*)d_in[18];
  // d_in[19] = length (always 768 here, compile-time)
  const int* ctx_mask      = (const int*)d_in[20];
  const int* tgt_mask      = (const int*)d_in[21];
  (void)in_sizes; (void)n_in; (void)out_size; (void)ws_size;

  // --- workspace carve (~159 MB) ---
  char* w = (char*)d_ws;
  auto carve = [&](size_t bytes) -> char* {
    char* p = w; w += (bytes + 255) & ~size_t(255); return p;
  };
  int*            keep_idx = (int*)           carve((size_t)LKK * 4);
  unsigned short* embed_wT = (unsigned short*)carve((size_t)DIMD * DIMD * 2);
  unsigned short* wqkvT    = (unsigned short*)carve((size_t)4 * 3 * DIMD * DIMD * 2);
  unsigned short* woT      = (unsigned short*)carve((size_t)4 * DIMD * DIMD * 2);
  unsigned short* w1T      = (unsigned short*)carve((size_t)4 * DIMD * HIDD * 2);
  unsigned short* w2T      = (unsigned short*)carve((size_t)4 * HIDD * DIMD * 2);
  unsigned short* xctx     = (unsigned short*)carve((size_t)BQ * LKK * DIMD * 2);
  float*          hbuf     = (float*)         carve((size_t)ROWS * DIMD * 4);
  unsigned short* ybuf     = (unsigned short*)carve((size_t)ROWS * DIMD * 2);
  float*          qkv      = (float*)         carve((size_t)ROWS * 3 * DIMD * 4);
  unsigned short* aout     = (unsigned short*)carve((size_t)ROWS * DIMD * 2);
  unsigned short* hid      = (unsigned short*)carve((size_t)ROWS * HIDD * 2);

  const int TPB = 256;
  const size_t DD = (size_t)DIMD * DIMD;
  const size_t DH = (size_t)DIMD * HIDD;

  // --- weight prep: fp32 (K,N) -> bf16 (N,K) ---
  cvt_transpose<<<(DD + TPB - 1) / TPB, TPB, 0, stream>>>(embed_w, embed_wT, DIMD, DIMD);
  for (int l = 0; l < 4; ++l) {
    for (int j = 0; j < 3; ++j)
      cvt_transpose<<<(DD + TPB - 1) / TPB, TPB, 0, stream>>>(
          Wqkv + (size_t)(l * 3 + j) * DD, wqkvT + (size_t)(l * 3 + j) * DD, DIMD, DIMD);
    cvt_transpose<<<(DD + TPB - 1) / TPB, TPB, 0, stream>>>(Wo + (size_t)l * DD, woT + (size_t)l * DD, DIMD, DIMD);
    cvt_transpose<<<(DH + TPB - 1) / TPB, TPB, 0, stream>>>(W1 + (size_t)l * DH, w1T + (size_t)l * DH, DIMD, HIDD);
    cvt_transpose<<<(DH + TPB - 1) / TPB, TPB, 0, stream>>>(W2 + (size_t)l * DH, w2T + (size_t)l * DH, HIDD, DIMD);
  }

  // --- prologue ---
  build_keep<<<1, 32, 0, stream>>>(ctx_mask, keep_idx);
  build_xctx<<<(BQ * LKK * DIMD + TPB - 1) / TPB, TPB, 0, stream>>>(ctx_feature, pos_embed, keep_idx, xctx);
  {
    dim3 g(DIMD / 64, (BQ * LKK) / 16);  // embed GEMM -> temp in qkv buffer
    gemm_bf16_wmma<<<g, 32, 0, stream>>>(xctx, embed_wT, embed_b, nullptr, qkv, nullptr,
                                         BQ * LKK, DIMD, DIMD, 0);
  }
  replicate_ctx<<<(NSEQ * LKK * DIMD + TPB - 1) / TPB, TPB, 0, stream>>>(qkv, hbuf);
  build_tgt<<<(NSEQ * NTT * DIMD + TPB - 1) / TPB, TPB, 0, stream>>>(mask_tok, pos_embed, tgt_mask, hbuf);

  // --- transformer layers ---
  for (int l = 0; l < 4; ++l) {
    ln_rows_bf16<<<ROWS, 256, 0, stream>>>(hbuf, ln1w + l * DIMD, ln1b + l * DIMD, ybuf);
    {
      dim3 g((3 * DIMD) / 64, ROWS / 16);  // fused QKV GEMM, N = 2304
      gemm_bf16_wmma<<<g, 32, 0, stream>>>(ybuf, wqkvT + (size_t)l * 3 * DD, bqkv + (size_t)l * 3 * DIMD,
                                           nullptr, qkv, nullptr, ROWS, 3 * DIMD, DIMD, 0);
    }
    {
      dim3 g(SEQL, NHEADD, NSEQ);
      attention<<<g, 64, 0, stream>>>(qkv, aout);
    }
    {
      dim3 g(DIMD / 64, ROWS / 16);  // O-proj + residual (in place on hbuf)
      gemm_bf16_wmma<<<g, 32, 0, stream>>>(aout, woT + (size_t)l * DD, bo + (size_t)l * DIMD,
                                           hbuf, hbuf, nullptr, ROWS, DIMD, DIMD, 0);
    }
    ln_rows_bf16<<<ROWS, 256, 0, stream>>>(hbuf, ln2w + l * DIMD, ln2b + l * DIMD, ybuf);
    {
      dim3 g(HIDD / 64, ROWS / 16);  // MLP up + exact GELU, bf16 out
      gemm_bf16_wmma<<<g, 32, 0, stream>>>(ybuf, w1T + (size_t)l * DH, b1 + (size_t)l * HIDD,
                                           nullptr, nullptr, hid, ROWS, HIDD, DIMD, 1);
    }
    {
      dim3 g(DIMD / 64, ROWS / 16);  // MLP down + residual (in place on hbuf)
      gemm_bf16_wmma<<<g, 32, 0, stream>>>(hid, w2T + (size_t)l * DH, b2 + (size_t)l * DIMD,
                                           hbuf, hbuf, nullptr, ROWS, DIMD, HIDD, 0);
    }
  }

  // --- final LN on last 64 tokens ---
  final_ln<<<NSEQ * NTT, 256, 0, stream>>>(hbuf, ln_w, ln_b, (float*)d_out);
}
